// RoIAlign3D_17179869529
// MI455X (gfx1250) — compile-verified, matrix operand order checked by reading, not run on hardware
//
#include <hip/hip_runtime.h>

// RoIAlign3D for MI455X (gfx1250, wave32).
// Trilinear interpolation recast as GEMM with block-diagonal weights so the
// 8-corner weighted reduction runs on v_wmma_f32_16x16x32_f16 (f32 accum).
// Gather-bound problem: x-adjacent corner pairs fetched as single b64 loads
// in saddr(SGPR base)+voffset32 form; dual accumulators break the WMMA->WMMA
// RAW chain (no hazard NOPs); non-temporal stores keep L2 for the features.

typedef __attribute__((ext_vector_type(16))) _Float16 v16h;
typedef __attribute__((ext_vector_type(8)))  _Float16 v8h;
typedef __attribute__((ext_vector_type(8)))  float    v8f;
typedef __attribute__((ext_vector_type(2)))  float    v2f;
typedef __attribute__((ext_vector_type(4)))  int      v4i;

#define C_CH   256
#define DVOL   64
#define HVOL   64
#define WVOL   64
#define HW_    (HVOL * WVOL)
#define DHW_   (DVOL * HVOL * WVOL)
#define SPTS   343              // 7*7*7 sample points per ROI
#define SPAD   352              // padded to 22 tiles of 16
#define STILES 22
#define CTILES 16               // 256 channels / 16

// Unaligned (4B) 2-float global load; gfx1250 supports unaligned b64 access.
__device__ __forceinline__ v2f load2(const float* base, unsigned byteOff) {
    v2f r;
    __builtin_memcpy(&r, (const char*)base + byteOff, 8);
    return r;
}

__global__ __launch_bounds__(256)
void roialign3d_wmma(const float* __restrict__ feat,    // (B, C, 64,64,64) f32
                     const float* __restrict__ props,   // (B, N, 6) f32
                     float* __restrict__ out,           // (B*N, C, 343) f32
                     int nrois_per_batch) {
    // Per-ROI geometry in LDS: 4 pair-base offsets + 8 f16 weights / sample.
    // Pair q = (dz,dy) covers x-corners {dx=0, dx=1} at consecutive addresses.
    __shared__ int      s_poff[SPAD * 4];   // 5.5 KB (16B-aligned rows)
    __shared__ _Float16 s_w   [SPAD * 8];   // 5.5 KB (16B-aligned rows)

    const int roi = blockIdx.x;
    const int b   = roi / nrois_per_batch;
    const int tid = threadIdx.x;

    // ---------------- Phase 1: sample-point geometry --------------------
    const float* p = props + (size_t)roi * 6;
    const float x1 = p[0], y1 = p[1], z1 = p[2];
    const float x2 = p[3], y2 = p[4], z2 = p[5];
    const float sx = (x2 - x1) * (1.0f / 6.0f);
    const float sy = (y2 - y1) * (1.0f / 6.0f);
    const float sz = (z2 - z1) * (1.0f / 6.0f);

    for (int s = tid; s < SPAD; s += 256) {
        v4i orow = {0, 0, 0, 0};
        v8h wrow;
        #pragma unroll
        for (int k = 0; k < 8; ++k) wrow[k] = (_Float16)0.0f;

        if (s < SPTS) {
            const int zi = s / 49;
            const int yi = (s / 7) % 7;
            const int xi = s % 7;
            const float gz = z1 + sz * (float)zi;
            const float gy = y1 + sy * (float)yi;
            const float gx = x1 + sx * (float)xi;
            const float z0f = floorf(gz), y0f = floorf(gy), x0f = floorf(gx);
            const float wz = gz - z0f, wy = gy - y0f, wx = gx - x0f;
            const int z0 = (int)z0f, y0 = (int)y0f, x0 = (int)x0f;

            // Pair base clamped so the b64 never leaves the row: covers
            // x = xb, xb+1 with xb <= 62.
            const int  xb   = min(max(x0, 0), WVOL - 2);
            const bool okx0 = (x0 >= 0) && (x0 < WVOL);
            const bool okx1 = (x0 + 1 >= 0) && (x0 + 1 < WVOL);

            #pragma unroll
            for (int q = 0; q < 4; ++q) {           // q = dz*2 + dy
                const int dz = q >> 1, dy = q & 1;
                const int zz = z0 + dz, yy = y0 + dy;
                const bool okzy = (zz >= 0) && (zz < DVOL) &&
                                  (yy >= 0) && (yy < HVOL);
                const int zc = min(max(zz, 0), DVOL - 1);
                const int yc = min(max(yy, 0), HVOL - 1);
                orow[q] = zc * HW_ + yc * WVOL + xb;

                const float wzy = (dz ? wz : 1.0f - wz) *
                                  (dy ? wy : 1.0f - wy) *
                                  (okzy ? 1.0f : 0.0f);
                float w0 = wzy * (1.0f - wx) * (okx0 ? 1.0f : 0.0f);
                float w1 = wzy * wx          * (okx1 ? 1.0f : 0.0f);
                if (x0 >= WVOL - 1) { w1 = w0; w0 = 0.0f; }  // x=63 -> slot 1
                wrow[q * 2 + 0] = (_Float16)w0;
                wrow[q * 2 + 1] = (_Float16)w1;
            }
        }
        // Padding rows keep zero weights -> zero output rows (never stored).
        reinterpret_cast<v4i*>(s_poff)[s] = orow;   // ds_store_b128
        reinterpret_cast<v8h*>(s_w)[s]    = wrow;   // ds_store_b128
    }
    __syncthreads();

    // ---------------- Phase 2: WMMA tiles -------------------------------
    const int wave = tid >> 5;
    const int lane = tid & 31;
    const int l16  = lane & 15;              // A-row / B-col / D-col stripe
    const int hi   = (lane >= 16) ? 1 : 0;   // half-wave K / M split
    const int hi8  = hi * 8;

    const float*   fb       = feat + (size_t)b * C_CH * DHW_;
    float*         out_roi  = out  + (size_t)roi * C_CH * SPTS;
    const unsigned laneBOff = (unsigned)(l16 * DHW_) * 4u;  // channel byte off

    const v4i* offv = reinterpret_cast<const v4i*>(s_poff);  // 1 v4i / sample
    const v8h* wv   = reinterpret_cast<const v8h*>(s_w);     // 1 v8h / sample

    // All control flow below is wave-uniform -> EXEC all-ones at each WMMA.
    for (int stile = wave; stile < STILES; stile += 8) {
        const int sbase = stile * 16;

        // This lane's row of 8 trilinear weights: single ds_load_b128.
        const v8h wrow = wv[sbase + l16];

        // A fragments (rows = local sample, K = (sample-in-group, corner)):
        // block-diagonal weights, built once per sample-tile via VALU selects.
        v16h Af[4];
        #pragma unroll
        for (int g = 0; g < 4; ++g) {
            v16h a;
            #pragma unroll
            for (int h = 0; h < 16; ++h) {
                const int K = (h < 8 ? h : h + 8) + hi8;  // ISA 16-bit A layout
                const int j = K >> 3;      // sample within group (0..3)
                const int k = K & 7;       // corner  (0..7)
                a[h] = (l16 == 4 * g + j) ? wrow[k] : (_Float16)0.0f;
            }
            Af[g] = a;
        }

        // Gather byte-offsets for this lane (2 samples x 4 pairs per group):
        // ctile-invariant -> 32 u32 VGPRs, loaded via 8x ds_load_b128.
        // Max value: 15*DHW*4 + (DHW-1)*4 < 2^28, fits u32.
        unsigned bidx[32];
        #pragma unroll
        for (int g = 0; g < 4; ++g) {
            const int sjA = sbase + 4 * g + hi;       // j = hi   (h 0..7)
            const int sjB = sjA + 2;                  // j = hi+2 (h 8..15)
            const v4i oA = offv[sjA];
            const v4i oB = offv[sjB];
            #pragma unroll
            for (int q = 0; q < 4; ++q) {
                bidx[g * 8 + q]     = laneBOff + (unsigned)oA[q] * 4u;
                bidx[g * 8 + 4 + q] = laneBOff + (unsigned)oB[q] * 4u;
            }
        }

        const bool full = (sbase + 16 <= SPTS);  // stiles 0..20: no ragged edge

        for (int ctile = 0; ctile < CTILES; ++ctile) {
            // Wave-uniform base (SGPR pair) -> saddr + voffset32 gathers.
            const float* fb_u = fb + (size_t)(ctile * 16) * DHW_;

            // Opaque zero, pinned in-loop: each bidx+czero is a fresh 32-bit
            // value whose zext feeds the load -> ISel folds saddr+voffset32;
            // bidx itself is never rewritten, so no preservation copies.
            unsigned czero;
            asm volatile("v_mov_b32 %0, 0" : "=v"(czero));

            // Dual accumulators: consecutive WMMAs write disjoint D registers
            // -> no WMMA->WMMA RAW hazard NOPs; chain depth halved.
            v8f acc0 = {}, acc1 = {};
            #pragma unroll
            for (int g = 0; g < 4; ++g) {
                // 8x global b64 gathers: each covers both x-corners of a
                // (dz,dy) pair for this lane's channel.
                v2f pr[8];
                #pragma unroll
                for (int q = 0; q < 8; ++q)
                    pr[q] = load2(fb_u, bidx[g * 8 + q] + czero);

                // B fragment: K = (sample, corner) with corner k = q*2 + dx.
                v16h bf;
                #pragma unroll
                for (int h = 0; h < 16; ++h) {
                    const int k = h & 7;                   // corner index
                    const float v = (h < 8) ? pr[k >> 1][k & 1]
                                            : pr[4 + (k >> 1)][k & 1];
                    bf[h] = (_Float16)v;
                }
                // D(16 samples x 16 channels) += A(16x32) * B(32x16)
                if (g & 1)
                    acc1 = __builtin_amdgcn_wmma_f32_16x16x32_f16(
                            false, Af[g], false, bf, (short)0, acc1,
                            false, false);
                else
                    acc0 = __builtin_amdgcn_wmma_f32_16x16x32_f16(
                            false, Af[g], false, bf, (short)0, acc0,
                            false, false);
            }
            const v8f acc = acc0 + acc1;   // 8 v_add_f32 (fills hazard slots)

            // Store: lane holds channel ctile*16+l16, VGPR v -> sample
            // sbase+hi8+v (8 consecutive floats). Non-temporal: output is
            // write-once, keep L2 for the gathered feature volume.
            float* orow = out_roi + (size_t)(ctile * 16 + l16) * SPTS
                                  + sbase + hi8;
            if (full) {
                #pragma unroll
                for (int v = 0; v < 8; ++v)
                    __builtin_nontemporal_store(acc[v], orow + v);
            } else {
                #pragma unroll
                for (int v = 0; v < 8; ++v)
                    if (sbase + hi8 + v < SPTS)
                        __builtin_nontemporal_store(acc[v], orow + v);
            }
        }
    }
}

extern "C" void kernel_launch(void* const* d_in, const int* in_sizes, int n_in,
                              void* d_out, int out_size, void* d_ws, size_t ws_size,
                              hipStream_t stream) {
    const float* feat  = (const float*)d_in[0];   // (B, 256, 64,64,64) f32
    const float* props = (const float*)d_in[1];   // (B, N, 6) f32
    float*       out   = (float*)d_out;           // (B*N, 256, 343) f32

    const int total_rois = in_sizes[1] / 6;                 // B*N = 1024
    const int batches    = in_sizes[0] / (C_CH * DHW_);     // B   = 2
    const int nper       = total_rois / batches;            // N   = 512

    roialign3d_wmma<<<dim3(total_rois), dim3(256), 0, stream>>>(
        feat, props, out, nper);
}
